// InfoNCELoss_28192165331072
// MI455X (gfx1250) — matrix operand order repaired and checked by reading, not compiled, not run on hardware
//
#include <hip/hip_runtime.h>

typedef __attribute__((ext_vector_type(16))) _Float16     v16h;
typedef __attribute__((ext_vector_type(8)))  float        v8f;
typedef __attribute__((ext_vector_type(4)))  unsigned int u32x4;
typedef __attribute__((ext_vector_type(4)))  unsigned int tdm_g0_t;
typedef __attribute__((ext_vector_type(8)))  int          tdm_g1_t;
typedef __attribute__((ext_vector_type(4)))  int          tdm_g2_t;
typedef __attribute__((ext_vector_type(8)))  int          tdm_g4_t;

#define B_N 8192
#define D_K 512
#define INV_T   14.285714285714286f   /* 1 / 0.07 */
#define TTHRESH 5.0e6f
#define BLK_COLS 64                   /* columns staged per block            */
#define NBLK (B_N / BLK_COLS)         /* 128 blocks                          */
#define BSTRIDE 520                   /* padded halves per staged row        */

#if defined(__has_builtin)
#if __has_builtin(__builtin_amdgcn_tensor_load_to_lds) && \
    __has_builtin(__builtin_amdgcn_s_wait_tensorcnt)
#define HAVE_TDM 1
#endif
#endif
#ifndef HAVE_TDM
#define HAVE_TDM 0
#endif

union HFrag { v16h v; u32x4 u[2]; };

// ---------------------------------------------------------------------------
// Kernel 0: f32 -> f16 convert of the descriptor matrix; zero global accums.
// ---------------------------------------------------------------------------
__global__ void convert_kernel(const float* __restrict__ in,
                               _Float16* __restrict__ out,
                               float* gTotal, unsigned int* gCount, int n) {
    int idx = blockIdx.x * blockDim.x + threadIdx.x;
    if (idx == 0) { *gTotal = 0.0f; *gCount = 0u; }
    if (idx < n) out[idx] = (_Float16)in[idx];
}

#if HAVE_TDM
// Issue one 2D TDM load: tile = 64 rows x 512 f16 (row-major, contiguous in
// the 8192x512 tensor), into LDS at lds_off with a 16B pad after every
// 1024B row (=> 1040B LDS row stride = BSTRIDE halves).
__device__ __forceinline__ void tdm_stage(const _Float16* tile_base,
                                          unsigned lds_off) {
    unsigned long long ga = (unsigned long long)(uintptr_t)tile_base;
    tdm_g0_t g0;
    g0.x = 1u;                                        // count=1, user D#
    g0.y = lds_off;                                   // lds_addr (bytes)
    g0.z = (unsigned)(ga & 0xFFFFFFFFu);              // global_addr[31:0]
    g0.w = (unsigned)((ga >> 32) & 0x01FFFFFFu)       // global_addr[56:32]
           | 0x80000000u;                             // type = 2 ("image")
    tdm_g1_t g1;
    g1[0] = (int)((1u << 16)        /* data_size = 2 bytes            */
                | (1u << 20)        /* pad_enable                     */
                | (7u << 22)        /* pad_interval: 256 DWORDs=1024B */
                | (3u << 25));      /* pad_amount:   4 DWORDs = 16B   */
    g1[1] = (int)(512u << 16);      // abar_addr=0 | tensor_dim0.lo16 (512)
    g1[2] = (int)(8192u << 16);     // tensor_dim0.hi16=0 | tensor_dim1.lo16
    g1[3] = (int)(512u << 16);      // tensor_dim1.hi16=0 | tile_dim0 = 512
    g1[4] = 64;                     // tile_dim1 = 64 rows | tile_dim2 = 0
    g1[5] = 512;                    // tensor_dim0_stride.lo32 = 512 elems
    g1[6] = 0;                      // stride0.hi16 | tensor_dim1_stride.lo16
    g1[7] = 0;
    tdm_g2_t gz4 = {0, 0, 0, 0};            // 2D tensor: groups 2/3 unused
    tdm_g4_t gz8 = {0, 0, 0, 0, 0, 0, 0, 0};
    __builtin_amdgcn_tensor_load_to_lds(g0, g1, gz4, gz4, gz8, 0);
}
#else
__device__ __forceinline__ void coop_copy(const _Float16* __restrict__ hmat,
                                          int blk, _Float16* dst) {
    const _Float16* src = hmat + (size_t)blk * BLK_COLS * D_K;
    for (int c = threadIdx.x; c < BLK_COLS * (D_K / 8); c += 256) {
        int r = c / (D_K / 8), kc = c % (D_K / 8);
        *(u32x4*)(dst + r * BSTRIDE + kc * 8) =
            *(const u32x4*)(src + r * D_K + kc * 8);
    }
}
#endif

// ---------------------------------------------------------------------------
// Kernel 1: fused GEMM (WMMA f16 -> f32) + masked streaming exp-sums.
// WG = 256 threads (8 waves) owns 64 rows. B columns staged per-WG into a
// double-buffered LDS region by the Tensor Data Mover, shared by all waves.
// wave w: row sub-block sb = w%4, column pair h = w/4 within each block.
// ---------------------------------------------------------------------------
__launch_bounds__(256)
__global__ void infonce_kernel(const _Float16* __restrict__ hmat,
                               const int*      __restrict__ seq,
                               const float*    __restrict__ times,
                               float* gTotal, unsigned int* gCount) {
    extern __shared__ _Float16 smem_h[];
    _Float16* sB   = smem_h;                                  // [2][64*BSTRIDE]
    float*    sAll = (float*)(smem_h + 2 * BLK_COLS * BSTRIDE);
    float*    sPos = sAll + 64;

    const int tid     = threadIdx.x;
    const int wave    = tid >> 5;
    const int lane    = tid & 31;
    const int sb      = wave & 3;
    const int h       = wave >> 2;
    const int nl      = lane & 15;
    const int khalf   = lane >> 4;
    const int rowBase = blockIdx.x * 64;

    if (tid < 64) { sAll[tid] = 0.0f; sPos[tid] = 0.0f; }

    // --- Preload this wave's A fragments (16 rows x 512 K) into registers.
    // 16-bit A 16x32 layout: lane holds row lane%16; lo chunk K=+8*khalf..+7,
    // hi chunk K=+16+8*khalf..+7.
    const _Float16* aRow = hmat + (size_t)(rowBase + sb * 16 + nl) * D_K;
    HFrag aF[16];
    #pragma unroll
    for (int kb = 0; kb < 16; ++kb) {
        aF[kb].u[0] = *(const u32x4*)(aRow + kb * 32 + 8 * khalf);
        aF[kb].u[1] = *(const u32x4*)(aRow + kb * 32 + 16 + 8 * khalf);
    }

    // --- Row metadata for the 8 C/D rows this lane holds.
    int   seq_i[8];  float time_i[8];  int row_i[8];
    #pragma unroll
    for (int g = 0; g < 8; ++g) {
        int i     = rowBase + sb * 16 + g + 8 * khalf;
        row_i[g]  = i;
        seq_i[g]  = seq[i];
        time_i[g] = times[i];
    }

    float sumAll[8], sumPos[8];
    #pragma unroll
    for (int g = 0; g < 8; ++g) { sumAll[g] = 0.0f; sumPos[g] = 0.0f; }

#if HAVE_TDM
    const unsigned ldsOff0 = (unsigned)(uintptr_t)sB;
    const unsigned bufBytes = BLK_COLS * BSTRIDE * sizeof(_Float16);
    if (wave == 0)
        tdm_stage(hmat, ldsOff0);                      // stage block 0 -> buf 0
#else
    coop_copy(hmat, 0, sB);
#endif

    for (int b = 0; b < NBLK; ++b) {
        const int buf = b & 1;
#if HAVE_TDM
        if (wave == 0)
            __builtin_amdgcn_s_wait_tensorcnt(0);      // block b resident
#endif
        __syncthreads();                               // publish buf to all waves
#if HAVE_TDM
        if (wave == 0 && b + 1 < NBLK)                 // overlap next block DMA
            tdm_stage(hmat + (size_t)(b + 1) * BLK_COLS * D_K,
                      ldsOff0 + (unsigned)(buf ^ 1) * bufBytes);
#else
        if (b + 1 < NBLK)
            coop_copy(hmat, b + 1, sB + (size_t)(buf ^ 1) * BLK_COLS * BSTRIDE);
#endif

        // --- Compute: this wave handles column tiles h*2 and h*2+1 of block b.
        const _Float16* sBuf = sB + (size_t)buf * BLK_COLS * BSTRIDE;
        #pragma unroll
        for (int tt = 0; tt < 2; ++tt) {
            const int tIn = h * 2 + tt;
            const int j   = b * BLK_COLS + tIn * 16 + nl;
            const _Float16* bBase = sBuf + (tIn * 16 + nl) * BSTRIDE;

            v8f acc = {};
            #pragma unroll
            for (int kb = 0; kb < 16; ++kb) {
                HFrag bf;   // B 32x16: lo chunk K=+8*khalf..+7, hi chunk K=+16+8*khalf..+7
                bf.u[0] = *(const u32x4*)(bBase + kb * 32 + 8 * khalf);
                bf.u[1] = *(const u32x4*)(bBase + kb * 32 + 16 + 8 * khalf);
                acc = __builtin_amdgcn_wmma_f32_16x16x32_f16(
                          false, aF[kb].v, false, bf.v, (short)0, acc,
                          false, false);
            }

            const int   seq_j  = seq[j];
            const float time_j = times[j];
            #pragma unroll
            for (int g = 0; g < 8; ++g) {
                if (j == row_i[g]) continue;            // exclude diagonal
                float e = __expf(acc[g] * INV_T - INV_T); // bounded (0,1]
                sumAll[g] += e;
                if (seq_j == seq_i[g] && fabsf(time_j - time_i[g]) < TTHRESH)
                    sumPos[g] += e;
            }
        }
    }

    // --- Merge per-lane partials into per-row LDS accumulators.
    #pragma unroll
    for (int g = 0; g < 8; ++g) {
        int lr = sb * 16 + g + 8 * khalf;
        atomicAdd(&sAll[lr], sumAll[g]);
        atomicAdd(&sPos[lr], sumPos[g]);
    }
    __syncthreads();

    // --- Per-row loss: log(sum_all) - log(sum_pos); the +C offsets cancel.
    if (tid < 64) {
        float a = sAll[tid], p = sPos[tid];
        if (p > 0.0f) {                                // p>0 <=> num_pos>0
            atomicAdd(gTotal, __logf(a) - __logf(p));
            atomicAdd(gCount, 1u);
        }
    }
}

// ---------------------------------------------------------------------------
// Kernel 2: finalize scalar loss.
// ---------------------------------------------------------------------------
__global__ void finalize_kernel(const float* gTotal, const unsigned int* gCount,
                                float* out) {
    if (threadIdx.x == 0 && blockIdx.x == 0) {
        unsigned int c = *gCount;
        float denom = (float)(c > 0u ? c : 1u);
        out[0] = *gTotal / denom;
    }
}

// ---------------------------------------------------------------------------
extern "C" void kernel_launch(void* const* d_in, const int* in_sizes, int n_in,
                              void* d_out, int out_size, void* d_ws, size_t ws_size,
                              hipStream_t stream) {
    const float* desc  = (const float*)d_in[0];   // [8192, 512] f32, unit rows
    const int*   seq   = (const int*)d_in[1];     // [8192] i32
    const float* times = (const float*)d_in[2];   // [8192] f32
    float* out = (float*)d_out;

    _Float16* hmat = (_Float16*)d_ws;                              // 8 MB f16 copy
    char* wsb = (char*)d_ws;
    float* gTotal        = (float*)(wsb + (size_t)B_N * D_K * sizeof(_Float16));
    unsigned int* gCount = (unsigned int*)(gTotal + 1);

    const int n = B_N * D_K;
    convert_kernel<<<(n + 255) / 256, 256, 0, stream>>>(desc, hmat, gTotal, gCount, n);

    const size_t shmem = 2u * BLK_COLS * BSTRIDE * sizeof(_Float16)  // B buffers
                       + 2u * 64 * sizeof(float);                    // sAll/sPos
    infonce_kernel<<<B_N / 64, 256, shmem, stream>>>(hmat, seq, times,
                                                     gTotal, gCount);
    finalize_kernel<<<1, 64, 0, stream>>>(gTotal, gCount, out);
}
